// MMD_loss_274877907592
// MI455X (gfx1250) — matrix operand order repaired and checked by reading, not compiled, not run on hardware
//
#include <hip/hip_runtime.h>
#include <hip/hip_bf16.h>

// MMD loss, fused single-GEMM-pass implementation for gfx1250 (MI455X).
//   Phase Z: zero column-sum accumulators
//   Phase A: f32 -> bf16 convert + row squared-norms + column sums
//   Phase B: closed-form bandwidth  (sum(l2) = 2n*sum|x|^2 - 2|sum x|^2), zero out
//   Phase C: symmetric (lower-triangle, x2 off-diagonal) 64x64-per-wave bf16 WMMA
//            Gram tiles fused with the 5-scale exp kernel + signed reduction
//            (one v_exp_f32 per element: t + t^2 + t^4 + t^8 + t^16)

typedef __bf16 bf16_t;
typedef __attribute__((ext_vector_type(16))) __bf16 v16bf;
typedef __attribute__((ext_vector_type(8)))  float  v8f;

#define HALF_N   4096
#define N_TOTAL  8192
#define DIM      512
#define TILES    128            // 8192 / 64 : 64-row wave tiles per dimension
#define NPAIRS   (TILES * (TILES + 1) / 2)   // 8256 lower-triangle tile pairs

// workspace layout (bytes)
#define TB_BYTES   (N_TOTAL * DIM * 2)          // 8 MB  bf16 matrix
#define SQ_OFF     (TB_BYTES)                   // 8192 f32 row norms
#define COL_OFF    (SQ_OFF + N_TOTAL * 4)       // 512  f32 column sums
#define SCAL_OFF   (COL_OFF + DIM * 4)          // scalars (c2)

__global__ __launch_bounds__(512) void mmd_zero_kernel(float* __restrict__ colsum) {
    int t = threadIdx.x;
    if (t < DIM) colsum[t] = 0.f;
}

// 128 blocks x 256 threads; each wave owns 8 rows (wave32).
__global__ __launch_bounds__(256) void mmd_prep_kernel(
    const float* __restrict__ src, const float* __restrict__ tgt,
    bf16_t* __restrict__ tb, float* __restrict__ sq, float* __restrict__ colsum) {
    __shared__ float lcol[DIM];
    const int tid  = threadIdx.x;
    const int lane = tid & 31;
    const int wave = tid >> 5;
    for (int i = tid; i < DIM; i += 256) lcol[i] = 0.f;
    __syncthreads();

    float csum[16];
#pragma unroll
    for (int k = 0; k < 16; ++k) csum[k] = 0.f;

    const int rowBase = blockIdx.x * 64 + wave * 8;
#pragma unroll
    for (int r = 0; r < 8; ++r) {
        const int row = rowBase + r;
        const float* srcRow = (row < HALF_N) ? (src + (size_t)row * DIM)
                                             : (tgt + (size_t)(row - HALF_N) * DIM);
        bf16_t* dstRow = tb + (size_t)row * DIM;
        float s = 0.f;
#pragma unroll
        for (int k = 0; k < 16; ++k) {
            const int c = lane + 32 * k;
            float f = srcRow[c];
            dstRow[c] = (bf16_t)f;
            s = fmaf(f, f, s);
            csum[k] += f;
        }
#pragma unroll
        for (int off = 16; off > 0; off >>= 1) s += __shfl_xor(s, off, 32);
        if (lane == 0) sq[row] = s;
    }
#pragma unroll
    for (int k = 0; k < 16; ++k) atomicAdd(&lcol[lane + 32 * k], csum[k]);
    __syncthreads();
    for (int i = tid; i < DIM; i += 256) atomicAdd(&colsum[i], lcol[i]);
}

// single block: bandwidth scalar + zero output accumulator
__global__ __launch_bounds__(256) void mmd_bw_kernel(
    const float* __restrict__ sq, const float* __restrict__ colsum,
    float* __restrict__ scal, float* __restrict__ out) {
    __shared__ double red[256];
    const int tid = threadIdx.x;
    double s = 0.0;
    for (int i = tid; i < N_TOTAL; i += 256) s += (double)sq[i];
    double c = 0.0;
    for (int d = tid; d < DIM; d += 256) { double v = (double)colsum[d]; c += v * v; }

    red[tid] = s;
    __syncthreads();
    for (int off = 128; off > 0; off >>= 1) {
        if (tid < off) red[tid] += red[tid + off];
        __syncthreads();
    }
    const double sumsq = red[0];
    __syncthreads();
    red[tid] = c;
    __syncthreads();
    for (int off = 128; off > 0; off >>= 1) {
        if (tid < off) red[tid] += red[tid + off];
        __syncthreads();
    }
    const double sumcol2 = red[0];

    if (tid == 0) {
        const double n = (double)N_TOTAL;
        const double sum_l2    = 2.0 * n * sumsq - 2.0 * sumcol2;
        const double bandwidth = sum_l2 / (n * n - n);     // pre /4 shift
        // largest scale = (bandwidth/4) * 2^4 = 4*bandwidth; fold log2(e) for exp2
        scal[0] = (float)(1.4426950408889634 / (4.0 * bandwidth));
        out[0]  = 0.f;
    }
}

__device__ __forceinline__ float mmd_tile_ep(const v8f& d, const float* sqA,
                                             float sqn, float c2) {
    float local = 0.f;
#pragma unroll
    for (int v = 0; v < 8; ++v) {
        float l2 = fmaxf(sqA[v] + sqn - 2.f * d[v], 0.f);
        float t  = __builtin_amdgcn_exp2f(-l2 * c2);     // exp(-l2 / (4*bw))
        float t2 = t * t, t4 = t2 * t2, t8 = t4 * t4, t16 = t8 * t8;
        local += t + t2 + t4 + t8 + t16;                 // all 5 bandwidth scales
    }
    return local;
}

// 1032 blocks x 256 threads; each wave computes one lower-triangle 64x64 tile
// (4x4 WMMA accumulators: 16 loads : 16 v_wmma_f32_16x16x32_bf16 per K-step).
__global__ __launch_bounds__(256) void mmd_main_kernel(
    const bf16_t* __restrict__ tb, const float* __restrict__ sq,
    const float* __restrict__ scal, float* __restrict__ out) {
    const int tid  = threadIdx.x;
    const int lane = tid & 31;
    const int h    = lane >> 4;       // half-wave (ISA fragment layouts)
    const int m    = lane & 15;
    const int w    = blockIdx.x * 8 + (tid >> 5);     // 0 .. NPAIRS-1

    // decode lower-triangle pair: i in [0,128), j <= i
    int ti = (int)((sqrtf(8.f * (float)w + 1.f) - 1.f) * 0.5f);
    while ((ti + 1) * (ti + 2) / 2 <= w) ++ti;
    while (ti * (ti + 1) / 2 > w) --ti;
    const int tj = w - ti * (ti + 1) / 2;

    const float c2 = scal[0];

    // A 16x32 bf16 fragment: lane m holds K = h*8..h*8+7 and 16+h*8..16+h*8+7
    const int aOff0 = h * 8;
    const int aOff1 = 16 + h * 8;
    // B 32x16 bf16 fragment: lane n holds K = h*8..h*8+7 and 16+h*8..16+h*8+7
    const int bOff  = h * 8;

    const bf16_t* pa[4];
    const bf16_t* pb[4];
#pragma unroll
    for (int p = 0; p < 4; ++p) {
        pa[p] = tb + (size_t)(ti * 64 + p * 16 + m) * DIM;
        pb[p] = tb + (size_t)(tj * 64 + p * 16 + m) * DIM;
    }

    v8f acc[4][4];
#pragma unroll
    for (int p = 0; p < 4; ++p)
#pragma unroll
        for (int q = 0; q < 4; ++q) acc[p][q] = (v8f){};

    for (int kk = 0; kk < DIM; kk += 32) {
        v16bf a[4], b[4];
#pragma unroll
        for (int p = 0; p < 4; ++p) {
            ((uint4*)&a[p])[0] = *(const uint4*)(pa[p] + kk + aOff0);
            ((uint4*)&a[p])[1] = *(const uint4*)(pa[p] + kk + aOff1);
            ((uint4*)&b[p])[0] = *(const uint4*)(pb[p] + kk + bOff);
            ((uint4*)&b[p])[1] = *(const uint4*)(pb[p] + kk + bOff + 16);
        }
#pragma unroll
        for (int p = 0; p < 4; ++p)
#pragma unroll
            for (int q = 0; q < 4; ++q)
                acc[p][q] = __builtin_amdgcn_wmma_f32_16x16x32_bf16(
                    false, a[p], false, b[q], (short)0, acc[p][q], false, false);
    }

    // epilogue: fused l2 -> 5-scale kernel sum -> signed accumulation
    float local = 0.f;
#pragma unroll
    for (int p = 0; p < 4; ++p) {
        float sqA[8];
#pragma unroll
        for (int v = 0; v < 8; ++v)
            sqA[v] = sq[ti * 64 + p * 16 + h * 8 + v];   // C layout: M = v + h*8
#pragma unroll
        for (int q = 0; q < 4; ++q) {
            const float sqn = sq[tj * 64 + q * 16 + m];
            local += mmd_tile_ep(acc[p][q], sqA, sqn, c2);
        }
    }

    // tile sign/weight are wave-uniform (4096 % 64 == 0):
    //   XX/YY -> +, XY/YX -> - ; off-diagonal tiles counted twice by symmetry
    const float sgn = ((ti < 64) == (tj < 64)) ? 1.f : -1.f;
    const float wgt = (ti == tj) ? 1.f : 2.f;
    local *= sgn * wgt * (1.0f / ((float)HALF_N * (float)HALF_N));

#pragma unroll
    for (int off = 16; off > 0; off >>= 1) local += __shfl_xor(local, off, 32);
    if (lane == 0) atomicAdd(out, local);
}

extern "C" void kernel_launch(void* const* d_in, const int* in_sizes, int n_in,
                              void* d_out, int out_size, void* d_ws, size_t ws_size,
                              hipStream_t stream) {
    const float* src = (const float*)d_in[0];
    const float* tgt = (const float*)d_in[1];
    float* out = (float*)d_out;

    char*   ws     = (char*)d_ws;               // needs ~8.43 MB
    bf16_t* tb     = (bf16_t*)ws;
    float*  sq     = (float*)(ws + SQ_OFF);
    float*  colsum = (float*)(ws + COL_OFF);
    float*  scal   = (float*)(ws + SCAL_OFF);

    mmd_zero_kernel<<<1, 512, 0, stream>>>(colsum);
    mmd_prep_kernel<<<N_TOTAL / 64, 256, 0, stream>>>(src, tgt, tb, sq, colsum);
    mmd_bw_kernel<<<1, 256, 0, stream>>>(sq, colsum, scal, out);
    mmd_main_kernel<<<NPAIRS / 8, 256, 0, stream>>>(tb, sq, scal, out);
}
